// CausalSelfAttention_7911329759309
// MI455X (gfx1250) — matrix operand order repaired and checked by reading, not compiled, not run on hardware
//
#include <hip/hip_runtime.h>

typedef __attribute__((ext_vector_type(16))) _Float16 v16h;
typedef __attribute__((ext_vector_type(8)))  _Float16 v8h;
typedef __attribute__((ext_vector_type(8)))  float    v8f;

#define B_ 2
#define S_ 2048
#define E_ 1024
#define H_ 16
#define D_ 64
#define M_ (B_ * S_)   // 4096 rows in the projection GEMMs

__device__ __forceinline__ v8f wmma_f16(v16h a, v16h b, v8f c) {
    // D = A(16x32 f16) * B(32x16 f16) + C(16x16 f32)
    return __builtin_amdgcn_wmma_f32_16x16x32_f16(
        /*neg_a=*/false, a, /*neg_b=*/false, b,
        /*c_mod=*/(short)0, c, /*reuse_a=*/false, /*reuse_b=*/false);
}

// ---------------------------------------------------------------------------
// fp32 -> fp16 elementwise conversion (for x)
// ---------------------------------------------------------------------------
__global__ void f32_to_f16(const float* __restrict__ in,
                           _Float16* __restrict__ out, int n) {
    for (int i = blockIdx.x * blockDim.x + threadIdx.x; i < n;
         i += gridDim.x * blockDim.x)
        out[i] = (_Float16)in[i];
}

// ---------------------------------------------------------------------------
// fp32 -> fp16 conversion WITH transpose (for weights):
// in:  W  [K=E_][N=E_] row-major f32
// out: Wt [N=E_][K=E_] row-major f16   (so WMMA B-fragments are contiguous)
// 32x32 tiles through LDS. grid (E_/32, E_/32), block (32, 8).
// ---------------------------------------------------------------------------
__global__ __launch_bounds__(256)
void convert_transpose(const float* __restrict__ in, _Float16* __restrict__ out) {
    __shared__ _Float16 tile[32][33];
    const int k0 = blockIdx.x * 32;
    const int n0 = blockIdx.y * 32;
    #pragma unroll
    for (int j = threadIdx.y; j < 32; j += 8)
        tile[j][threadIdx.x] = (_Float16)in[(size_t)(k0 + j) * E_ + n0 + threadIdx.x];
    __syncthreads();
    #pragma unroll
    for (int j = threadIdx.y; j < 32; j += 8)
        out[(size_t)(n0 + j) * E_ + k0 + threadIdx.x] = tile[threadIdx.x][j];
}

// ---------------------------------------------------------------------------
// GEMM: C[M x N] = A[M x K] * W[K x N] + bias,  M=4096, N=K=1024 (f16 in),
// with W supplied TRANSPOSED (Wt[N][K]).
// mode 0: f32 row-major output (final projection -> d_out)
// mode 1: f16 head-split [B,H,S,D] (Q, K projections)
// mode 2: f16 head-split transposed [B,H,D,S] (V projection, for fast P@V)
// block = 128 threads (4 waves); each wave computes a 32(M) x 64(N) tile:
// per K-step: 2 A fragments + 4 B fragments -> 8 WMMAs, B frags reused 2x.
// ---------------------------------------------------------------------------
__global__ __launch_bounds__(128)
void gemm16(const _Float16* __restrict__ A, const _Float16* __restrict__ Wt,
            const float* __restrict__ bias, _Float16* __restrict__ outHS,
            float* __restrict__ outF, int mode) {
    const int lane = threadIdx.x & 31;
    const int w    = threadIdx.x >> 5;   // wave 0..3
    const int ln   = lane & 15;
    const int l16  = lane >> 4;
    const int mbase = blockIdx.x * 32;
    const int nwave = blockIdx.y * 256 + w * 64;

    v8f acc[2][4] = {};

    for (int kk = 0; kk < E_; kk += 32) {
        // Two A fragments (rows mbase..+15 and mbase+16..+31), contiguous 16B loads
        union { v16h v; v8h h2[2]; } af[2];
        #pragma unroll
        for (int mb = 0; mb < 2; ++mb) {
            const _Float16* asrc = A + (size_t)(mbase + 16 * mb + ln) * E_ + kk;
            af[mb].h2[0] = *(const v8h*)(asrc + 16 * l16);
            af[mb].h2[1] = *(const v8h*)(asrc + 16 * l16 + 8);
        }

        if (kk + 32 < E_)
            __builtin_prefetch(Wt + (size_t)(nwave + ln) * E_ + kk + 32, 0, 1);

        // Four B fragments (one contiguous 32B load each), each feeds 2 WMMAs
        #pragma unroll
        for (int t = 0; t < 4; ++t) {
            const v16h bf = *(const v16h*)(Wt + (size_t)(nwave + 16 * t + ln) * E_
                                              + kk + 16 * l16);
            acc[0][t] = wmma_f16(af[0].v, bf, acc[0][t]);
            acc[1][t] = wmma_f16(af[1].v, bf, acc[1][t]);
        }
    }

    #pragma unroll
    for (int t = 0; t < 4; ++t) {
        const int nglob = nwave + 16 * t + ln;
        const float bv = bias[nglob];
        #pragma unroll
        for (int mb = 0; mb < 2; ++mb) {
            #pragma unroll
            for (int r = 0; r < 8; ++r) {
                const int m = mbase + 16 * mb + r + 8 * l16;  // C layout: M = r + 8*l16
                const float val = acc[mb][t][r] + bv;
                if (mode == 1) {
                    const int b = m >> 11, s = m & (S_ - 1);
                    const int hd = nglob >> 6, d = nglob & 63;
                    outHS[(((size_t)(b * H_ + hd)) * S_ + s) * D_ + d] = (_Float16)val;
                } else if (mode == 2) {
                    const int b = m >> 11, s = m & (S_ - 1);
                    const int hd = nglob >> 6, d = nglob & 63;
                    outHS[(((size_t)(b * H_ + hd)) * D_ + d) * S_ + s] = (_Float16)val;
                } else {
                    outF[(size_t)m * E_ + nglob] = val;
                }
            }
        }
    }
}

// ---------------------------------------------------------------------------
// Flash attention (causal, online softmax). Q/K f16 [B,H,S,D]; V f16 [B,H,D,S].
// Output f16 written as [B,S,E] ( == [B,S,H,D] ) for the final GEMM.
// block = 128 threads (4 waves); each wave owns 16 query rows; grid covers
// (S/64) x (B*H). Keys processed in chunks of 32; the SxS score matrix never
// touches HBM (saves ~1 GB of traffic vs. materializing it).
// ---------------------------------------------------------------------------
__global__ __launch_bounds__(128)
void flash_attn(const _Float16* __restrict__ Q, const _Float16* __restrict__ K,
                const _Float16* __restrict__ Vt, _Float16* __restrict__ O) {
    __shared__ __align__(16) _Float16 sP[4][16 * 32];   // wave-private P tiles

    const int lane = threadIdx.x & 31;
    const int w    = threadIdx.x >> 5;
    const int ln   = lane & 15;
    const int l16  = lane >> 4;
    const int bh   = blockIdx.y;              // 0..31
    const int b    = bh >> 4, hd = bh & 15;
    const int sq0  = blockIdx.x * 64 + w * 16;

    const _Float16* Qb  = Q  + (size_t)bh * S_ * D_;
    const _Float16* Kb  = K  + (size_t)bh * S_ * D_;
    const _Float16* Vtb = Vt + (size_t)bh * D_ * S_;

    // Q fragments: 16 rows x D=64 -> two 16x32 A fragments (loaded once)
    union { v16h v; v8h h2[2]; } aq[2];
    #pragma unroll
    for (int c = 0; c < 2; ++c) {
        const _Float16* qsrc = Qb + (size_t)(sq0 + ln) * D_ + 32 * c;
        aq[c].h2[0] = *(const v8h*)(qsrc + 16 * l16);
        aq[c].h2[1] = *(const v8h*)(qsrc + 16 * l16 + 8);
    }

    float mrow[8], lrow[8];
    v8f acc[4] = {};
    #pragma unroll
    for (int r = 0; r < 8; ++r) { mrow[r] = -1e30f; lrow[r] = 0.0f; }

    const float scale = 0.125f;   // 1/sqrt(64)
    const int kend = sq0 + 16;    // causal horizon for this wave's 16 rows

    for (int kk = 0; kk < kend; kk += 32) {
        // ---- scores: two 16x16 tiles over this 32-key chunk ----
        v8f s0 = {}, s1 = {};
        #pragma unroll
        for (int c = 0; c < 2; ++c) {
            const v16h bk0 = *(const v16h*)(Kb + (size_t)(kk + ln) * D_ + 32 * c + 16 * l16);
            const v16h bk1 = *(const v16h*)(Kb + (size_t)(kk + 16 + ln) * D_ + 32 * c + 16 * l16);
            s0 = wmma_f16(aq[c].v, bk0, s0);
            s1 = wmma_f16(aq[c].v, bk1, s1);
        }

        // ---- mask + online softmax; stage P into LDS as f16 ----
        #pragma unroll
        for (int r = 0; r < 8; ++r) {
            const int qrow = sq0 + r + 8 * l16;
            float x0 = s0[r] * scale, x1 = s1[r] * scale;
            if (kk + ln > qrow)      x0 = -1e30f;
            if (kk + 16 + ln > qrow) x1 = -1e30f;

            float vmax = fmaxf(x0, x1);
            vmax = fmaxf(vmax, __shfl_xor(vmax, 1, 32));
            vmax = fmaxf(vmax, __shfl_xor(vmax, 2, 32));
            vmax = fmaxf(vmax, __shfl_xor(vmax, 4, 32));
            vmax = fmaxf(vmax, __shfl_xor(vmax, 8, 32));

            const float mnew = fmaxf(mrow[r], vmax);
            const float corr = __expf(mrow[r] - mnew);
            const float p0 = __expf(x0 - mnew);
            const float p1 = __expf(x1 - mnew);
            float rs = p0 + p1;
            rs += __shfl_xor(rs, 1, 32);
            rs += __shfl_xor(rs, 2, 32);
            rs += __shfl_xor(rs, 4, 32);
            rs += __shfl_xor(rs, 8, 32);
            lrow[r] = lrow[r] * corr + rs;
            mrow[r] = mnew;
            #pragma unroll
            for (int t = 0; t < 4; ++t) acc[t][r] *= corr;

            sP[w][(r + 8 * l16) * 32 + ln]      = (_Float16)p0;
            sP[w][(r + 8 * l16) * 32 + 16 + ln] = (_Float16)p1;
        }

        // ---- regather P in A-fragment layout: two contiguous 16B LDS loads ----
        // (same-wave DS ops complete in order, so no barrier needed)
        union { v16h v; v8h h2[2]; } pf;
        pf.h2[0] = *(const v8h*)&sP[w][ln * 32 + 16 * l16];
        pf.h2[1] = *(const v8h*)&sP[w][ln * 32 + 16 * l16 + 8];

        // ---- P(16x32) @ V(32x64): V transposed -> contiguous B fragments ----
        #pragma unroll
        for (int t = 0; t < 4; ++t) {
            const v16h bv = *(const v16h*)(Vtb + (size_t)(16 * t + ln) * S_
                                               + kk + 16 * l16);
            acc[t] = wmma_f16(pf.v, bv, acc[t]);
        }
    }

    // ---- normalize, store attention output as [B,S,H,D] f16 ----
    #pragma unroll
    for (int r = 0; r < 8; ++r) {
        const float inv = 1.0f / lrow[r];
        const int sq = sq0 + r + 8 * l16;
        _Float16* dst = O + ((size_t)(b * S_ + sq)) * E_ + hd * D_;
        #pragma unroll
        for (int t = 0; t < 4; ++t)
            dst[16 * t + ln] = (_Float16)(acc[t][r] * inv);
    }
}

// ---------------------------------------------------------------------------
// Host-side orchestration
// ---------------------------------------------------------------------------
extern "C" void kernel_launch(void* const* d_in, const int* in_sizes, int n_in,
                              void* d_out, int out_size, void* d_ws, size_t ws_size,
                              hipStream_t stream) {
    const float* x  = (const float*)d_in[0];
    const float* Wq = (const float*)d_in[1];
    const float* bq = (const float*)d_in[2];
    const float* Wk = (const float*)d_in[3];
    const float* bk = (const float*)d_in[4];
    const float* Wv = (const float*)d_in[5];
    const float* bv = (const float*)d_in[6];
    const float* Wo = (const float*)d_in[7];
    const float* bo = (const float*)d_in[8];
    float* out = (float*)d_out;

    // workspace carve-up (all fp16)
    _Float16* ws  = (_Float16*)d_ws;
    _Float16* xh  = ws;                        // 4096*1024
    _Float16* wqt = xh  + (size_t)M_ * E_;     // 1024*1024 (transposed f16)
    _Float16* wkt = wqt + (size_t)E_ * E_;
    _Float16* wvt = wkt + (size_t)E_ * E_;
    _Float16* wot = wvt + (size_t)E_ * E_;
    _Float16* Qh  = wot + (size_t)E_ * E_;     // [B,H,S,D]
    _Float16* Kh  = Qh  + (size_t)M_ * E_;     // [B,H,S,D]
    _Float16* Vth = Kh  + (size_t)M_ * E_;     // [B,H,D,S]
    _Float16* Ah  = Vth + (size_t)M_ * E_;     // attn out [B,S,E]

    const int nX = M_ * E_;
    f32_to_f16<<<(nX + 255) / 256, 256, 0, stream>>>(x, xh, nX);

    dim3 gt(E_ / 32, E_ / 32);   // 32 x 32 tiles
    dim3 bt(32, 8);
    convert_transpose<<<gt, bt, 0, stream>>>(Wq, wqt);
    convert_transpose<<<gt, bt, 0, stream>>>(Wk, wkt);
    convert_transpose<<<gt, bt, 0, stream>>>(Wv, wvt);
    convert_transpose<<<gt, bt, 0, stream>>>(Wo, wot);

    dim3 gg(M_ / 32, E_ / 256);  // 128 x 4 blocks, 4 waves each
    gemm16<<<gg, 128, 0, stream>>>(xh, wqt, bq, Qh,  nullptr, 1);
    gemm16<<<gg, 128, 0, stream>>>(xh, wkt, bk, Kh,  nullptr, 1);
    gemm16<<<gg, 128, 0, stream>>>(xh, wvt, bv, Vth, nullptr, 2);

    dim3 ga(S_ / 64, B_ * H_);   // 32 x 32 blocks
    flash_attn<<<ga, 128, 0, stream>>>(Qh, Kh, Vth, Ah);

    gemm16<<<gg, 128, 0, stream>>>(Ah, wot, bo, nullptr, out, 0);
}
